// TransformerBlock_32727650795902
// MI455X (gfx1250) — compile-verified, matrix operand order checked by reading, not run on hardware
//
#include <hip/hip_runtime.h>
#include <hip/hip_bf16.h>
#include <math.h>

typedef __attribute__((ext_vector_type(16))) __bf16 v16bf;
typedef __attribute__((ext_vector_type(8)))  __bf16 v8bf;
typedef __attribute__((ext_vector_type(8)))  float  v8f;

union Frag { v16bf v; v8bf h[2]; };

#define B_  2
#define S_  2048
#define D_  768
#define H_  12
#define HD_ 64
#define DFF_ 3072
#define M_  (B_ * S_)   // 4096

// ---------------------------------------------------------------------------
// fp32 -> bf16 elementwise convert
// ---------------------------------------------------------------------------
__global__ __launch_bounds__(256) void cvt_bf16_kernel(const float* __restrict__ in,
                                                       __bf16* __restrict__ out, int n) {
  int i = blockIdx.x * 256 + threadIdx.x;
  if (i < n) out[i] = (__bf16)in[i];
}

// ---------------------------------------------------------------------------
// fp32 (R x C) -> bf16 transposed (C x R)
// ---------------------------------------------------------------------------
__global__ __launch_bounds__(256) void transpose_bf16_kernel(const float* __restrict__ in,
                                                             __bf16* __restrict__ out,
                                                             int R, int C) {
  __shared__ float t[32][33];
  int c0 = blockIdx.x * 32, r0 = blockIdx.y * 32;
  for (int i = threadIdx.y; i < 32; i += 8)
    t[i][threadIdx.x] = in[(size_t)(r0 + i) * C + c0 + threadIdx.x];
  __syncthreads();
  for (int i = threadIdx.y; i < 32; i += 8)
    out[(size_t)(c0 + i) * R + r0 + threadIdx.x] = (__bf16)t[threadIdx.x][i];
}

// ---------------------------------------------------------------------------
// Generic bf16 WMMA GEMM: C[M,N] = A[M,K] @ BT[N,K]^T + bias, with epilogues.
// Block: 256 threads = 8 waves, tile 128(M) x 64(N), K-step 32.
// Double-buffered (ping-pong) LDS: one barrier per K-step; global->reg
// prefetch of step i+1 overlaps the WMMAs of step i.
// epi: 1 = bf16 head-split [B,H,S,HD] (scale applied)
//      2 = bf16 V-transposed [B,H,HD,S]
//      3 = fp32 out = acc + bias + resid
//      4 = bf16 relu(acc + bias), row-major
// ---------------------------------------------------------------------------
__global__ __launch_bounds__(256) void gemm_bf16_kernel(
    const __bf16* __restrict__ A, const __bf16* __restrict__ BT,
    const float* __restrict__ bias, const float* __restrict__ resid,
    void* __restrict__ out, int M, int N, int K, int epi, float scale) {
  __shared__ __attribute__((aligned(16))) __bf16 lA[2][128 * 40];
  __shared__ __attribute__((aligned(16))) __bf16 lB[2][64 * 40];

  const int tid = threadIdx.x;
  const int m0 = blockIdx.y * 128, n0 = blockIdx.x * 64;
  const int wid = tid >> 5, lane = tid & 31;
  const int wm = wid >> 1, wn = wid & 1;
  const int hi = (lane >> 4) & 1, col = lane & 15;

  // per-thread staging slots (A tile: 512 v8bf chunks -> 2/thread; B: 1/thread)
  const int ar0 = tid >> 2,         ac0 = (tid & 3) * 8;
  const int ar1 = (256 + tid) >> 2, ac1 = (tid & 3) * 8;
  const int br  = tid >> 2,         bc  = (tid & 3) * 8;
  const __bf16* Ag0 = &A[(size_t)(m0 + ar0) * K + ac0];
  const __bf16* Ag1 = &A[(size_t)(m0 + ar1) * K + ac1];
  const __bf16* Bg  = &BT[(size_t)(n0 + br) * K + bc];

  v8f acc[2][2] = {};

  // prologue: stage K-step 0 into buffer 0
  v8bf rA0 = *(const v8bf*)Ag0;
  v8bf rA1 = *(const v8bf*)Ag1;
  v8bf rB0 = *(const v8bf*)Bg;
  *(v8bf*)&lA[0][ar0 * 40 + ac0] = rA0;
  *(v8bf*)&lA[0][ar1 * 40 + ac1] = rA1;
  *(v8bf*)&lB[0][br * 40 + bc]   = rB0;
  __syncthreads();

  const int nk = K >> 5;
  for (int ki = 0; ki < nk; ++ki) {
    const int cur = ki & 1;
    const bool more = (ki + 1) < nk;
    if (more) {  // global -> register prefetch of next K-step
      int ko = (ki + 1) << 5;
      rA0 = *(const v8bf*)(Ag0 + ko);
      rA1 = *(const v8bf*)(Ag1 + ko);
      rB0 = *(const v8bf*)(Bg + ko);
    }

    Frag af[2], bfr[2];
#pragma unroll
    for (int t = 0; t < 2; ++t) {
      int ra = (wm * 32 + t * 16 + col) * 40 + hi * 8;
      af[t].h[0] = *(const v8bf*)&lA[cur][ra];
      af[t].h[1] = *(const v8bf*)&lA[cur][ra + 16];
      int rb = (wn * 32 + t * 16 + col) * 40 + hi * 8;
      bfr[t].h[0] = *(const v8bf*)&lB[cur][rb];
      bfr[t].h[1] = *(const v8bf*)&lB[cur][rb + 16];
    }
#pragma unroll
    for (int i = 0; i < 2; ++i)
#pragma unroll
      for (int j = 0; j < 2; ++j)
        acc[i][j] = __builtin_amdgcn_wmma_f32_16x16x32_bf16(
            false, af[i].v, false, bfr[j].v, (short)0, acc[i][j], false, false);

    if (more) {  // store next K-step into the other buffer (read-safe: barrier
                 // at end of previous iteration drained all readers of it)
      *(v8bf*)&lA[cur ^ 1][ar0 * 40 + ac0] = rA0;
      *(v8bf*)&lA[cur ^ 1][ar1 * 40 + ac1] = rA1;
      *(v8bf*)&lB[cur ^ 1][br * 40 + bc]   = rB0;
      __syncthreads();
    }
  }

  // epilogue
#pragma unroll
  for (int i = 0; i < 2; ++i)
#pragma unroll
    for (int j = 0; j < 2; ++j) {
      int mt = m0 + wm * 32 + i * 16;
      int nt = n0 + wn * 32 + j * 16;
      int cg = nt + col;
      float bb = bias ? bias[cg] : 0.0f;
      if (epi == 2) {
        // V transposed store: [B,H,HD,S]; rows mt+r+8*hi are s-contiguous
        int bI = mt >> 11, hh = cg >> 6, hd = cg & 63;
        int sbase = (mt & 2047) + 8 * hi;
        v8bf pk;
#pragma unroll
        for (int r = 0; r < 8; ++r) pk[r] = (__bf16)(acc[i][j][r] + bb);
        *(v8bf*)&((__bf16*)out)[(((size_t)bI * H_ + hh) * HD_ + hd) * S_ + sbase] = pk;
      } else {
#pragma unroll
        for (int r = 0; r < 8; ++r) {
          int rg = mt + r + 8 * hi;
          float v = acc[i][j][r] + bb;
          if (epi == 1) {
            int bI = rg >> 11, s = rg & 2047, hh = cg >> 6, hd = cg & 63;
            ((__bf16*)out)[(((size_t)bI * H_ + hh) * S_ + s) * HD_ + hd] = (__bf16)(v * scale);
          } else if (epi == 3) {
            ((float*)out)[(size_t)rg * N + cg] = v + resid[(size_t)rg * N + cg];
          } else {  // epi == 4: relu bf16
            ((__bf16*)out)[(size_t)rg * N + cg] = (__bf16)(v > 0.0f ? v : 0.0f);
          }
        }
      }
    }
}

// ---------------------------------------------------------------------------
// Flash attention with ALiBi + causal mask.
// Block = 128 threads (4 waves); each wave owns 16 q rows; keys in 64-chunks.
// Q,K: bf16 [B,H,S,HD] (Q pre-scaled by 1/8). Vt: bf16 [B,H,HD,S].
// ---------------------------------------------------------------------------
__device__ inline float half16_max(float v) {
  v = fmaxf(v, __shfl_xor(v, 1, 32));
  v = fmaxf(v, __shfl_xor(v, 2, 32));
  v = fmaxf(v, __shfl_xor(v, 4, 32));
  v = fmaxf(v, __shfl_xor(v, 8, 32));
  return v;
}
__device__ inline float half16_sum(float v) {
  v += __shfl_xor(v, 1, 32);
  v += __shfl_xor(v, 2, 32);
  v += __shfl_xor(v, 4, 32);
  v += __shfl_xor(v, 8, 32);
  return v;
}

__global__ __launch_bounds__(128) void attn_kernel(
    const __bf16* __restrict__ Q, const __bf16* __restrict__ Kmat,
    const __bf16* __restrict__ Vt, const float* __restrict__ alibi,
    __bf16* __restrict__ O) {
  __shared__ __attribute__((aligned(16))) __bf16 lp[4 * 16 * 72];

  const int tid = threadIdx.x, wid = tid >> 5, lane = tid & 31;
  const int hi = (lane >> 4) & 1, col = lane & 15;
  const int bh = blockIdx.x >> 5;          // 32 q-blocks per (b,h)
  const int qblk = blockIdx.x & 31;
  const int b = bh / H_, h = bh % H_;
  const int qw = qblk * 64 + wid * 16;     // this wave's first q row

  const size_t headQK = ((size_t)b * H_ + h) * S_ * HD_;
  const size_t headV = ((size_t)b * H_ + h) * HD_ * S_;
  const float* alH = alibi + (size_t)h * S_ * S_;

  // Q fragments for this wave's 16 rows (K-dim = 64 -> 2 frags), live all loop
  Frag qf[2];
  const __bf16* qrow = Q + headQK + (size_t)(qw + col) * HD_;
#pragma unroll
  for (int kf = 0; kf < 2; ++kf) {
    qf[kf].h[0] = *(const v8bf*)&qrow[kf * 32 + hi * 8];
    qf[kf].h[1] = *(const v8bf*)&qrow[kf * 32 + hi * 8 + 16];
  }

  float mrow[8], lrow[8];
  v8f Oacc[4] = {};
#pragma unroll
  for (int r = 0; r < 8; ++r) { mrow[r] = -INFINITY; lrow[r] = 0.0f; }

  __bf16* pbuf = &lp[wid * 16 * 72];

  for (int kb = 0; kb < qw + 16; kb += 64) {
    float sc[4][8];
#pragma unroll
    for (int nt = 0; nt < 4; ++nt) {
      const __bf16* krow = Kmat + headQK + (size_t)(kb + nt * 16 + col) * HD_;
      Frag k0, k1;
      k0.h[0] = *(const v8bf*)&krow[hi * 8];
      k0.h[1] = *(const v8bf*)&krow[hi * 8 + 16];
      k1.h[0] = *(const v8bf*)&krow[32 + hi * 8];
      k1.h[1] = *(const v8bf*)&krow[32 + hi * 8 + 16];
      v8f s = {};
      s = __builtin_amdgcn_wmma_f32_16x16x32_bf16(false, qf[0].v, false, k0.v,
                                                  (short)0, s, false, false);
      s = __builtin_amdgcn_wmma_f32_16x16x32_bf16(false, qf[1].v, false, k1.v,
                                                  (short)0, s, false, false);
      int kg = kb + nt * 16 + col;
      const float* al = alH + kg;
      __builtin_prefetch(al + 64, 0, 1);
#pragma unroll
      for (int r = 0; r < 8; ++r) {
        int qg = qw + r + 8 * hi;
        float v = s[r] + al[(size_t)qg * S_];
        sc[nt][r] = (kg <= qg) ? v : -INFINITY;
      }
    }

    // online softmax row statistics (rows split across lane halves)
    float alpha[8];
#pragma unroll
    for (int r = 0; r < 8; ++r) {
      float mx = fmaxf(fmaxf(sc[0][r], sc[1][r]), fmaxf(sc[2][r], sc[3][r]));
      mx = half16_max(mx);
      float mnew = fmaxf(mrow[r], mx);
      alpha[r] = __expf(mrow[r] - mnew);
      mrow[r] = mnew;
      float rsum = 0.0f;
#pragma unroll
      for (int nt = 0; nt < 4; ++nt) {
        float p = __expf(sc[nt][r] - mnew);
        sc[nt][r] = p;
        rsum += p;
      }
      rsum = half16_sum(rsum);
      lrow[r] = lrow[r] * alpha[r] + rsum;
    }

    // stage P (16x64) to LDS in row-major, re-read in A-fragment layout
#pragma unroll
    for (int nt = 0; nt < 4; ++nt)
#pragma unroll
      for (int r = 0; r < 8; ++r)
        pbuf[(r + 8 * hi) * 72 + nt * 16 + col] = (__bf16)sc[nt][r];
    asm volatile("" ::: "memory");  // keep DS store->load ordering (HW is in-order per wave)

    Frag pf[2];
#pragma unroll
    for (int kf = 0; kf < 2; ++kf) {
      int base = col * 72 + kf * 32 + hi * 8;
      pf[kf].h[0] = *(const v8bf*)&pbuf[base];
      pf[kf].h[1] = *(const v8bf*)&pbuf[base + 16];
    }

    // O = O*alpha + P @ V  (V pre-transposed: rows are hd, contiguous in s)
#pragma unroll
    for (int dt = 0; dt < 4; ++dt) {
#pragma unroll
      for (int r = 0; r < 8; ++r) Oacc[dt][r] *= alpha[r];
      const __bf16* vrow = Vt + headV + (size_t)(dt * 16 + col) * S_ + kb;
      Frag v0, v1;
      v0.h[0] = *(const v8bf*)&vrow[hi * 8];
      v0.h[1] = *(const v8bf*)&vrow[hi * 8 + 16];
      v1.h[0] = *(const v8bf*)&vrow[32 + hi * 8];
      v1.h[1] = *(const v8bf*)&vrow[32 + hi * 8 + 16];
      Oacc[dt] = __builtin_amdgcn_wmma_f32_16x16x32_bf16(false, pf[0].v, false, v0.v,
                                                         (short)0, Oacc[dt], false, false);
      Oacc[dt] = __builtin_amdgcn_wmma_f32_16x16x32_bf16(false, pf[1].v, false, v1.v,
                                                         (short)0, Oacc[dt], false, false);
    }
  }

  // normalize and store attn output as bf16 [B*S, D] (heads merged)
  float rl[8];
#pragma unroll
  for (int r = 0; r < 8; ++r) rl[r] = 1.0f / lrow[r];
#pragma unroll
  for (int dt = 0; dt < 4; ++dt)
#pragma unroll
    for (int r = 0; r < 8; ++r) {
      int qg = qw + r + 8 * hi;
      size_t off = ((size_t)b * S_ + qg) * D_ + h * HD_ + dt * 16 + col;
      O[off] = (__bf16)(Oacc[dt][r] * rl[r]);
    }
}

// ---------------------------------------------------------------------------
// Row LayerNorm over D=768; one block (256 thr) per row. Optional bf16 copy.
// ---------------------------------------------------------------------------
__global__ __launch_bounds__(256) void ln_kernel(const float* __restrict__ in,
                                                 const float* __restrict__ gamma,
                                                 const float* __restrict__ beta,
                                                 float* __restrict__ outf,
                                                 __bf16* __restrict__ outb) {
  __shared__ float rs[256], rq[256];
  const int row = blockIdx.x, tid = threadIdx.x;
  const float* p = in + (size_t)row * D_;
  float v0 = p[tid], v1 = p[tid + 256], v2 = p[tid + 512];
  rs[tid] = v0 + v1 + v2;
  rq[tid] = v0 * v0 + v1 * v1 + v2 * v2;
  __syncthreads();
  for (int o = 128; o > 0; o >>= 1) {
    if (tid < o) { rs[tid] += rs[tid + o]; rq[tid] += rq[tid + o]; }
    __syncthreads();
  }
  float mu = rs[0] * (1.0f / D_);
  float var = rq[0] * (1.0f / D_) - mu * mu;
  float rstd = rsqrtf(var + 1e-5f);
#pragma unroll
  for (int i = 0; i < 3; ++i) {
    int c = tid + i * 256;
    float x = (i == 0 ? v0 : (i == 1 ? v1 : v2));
    float y = (x - mu) * rstd * gamma[c] + beta[c];
    outf[(size_t)row * D_ + c] = y;
    if (outb) outb[(size_t)row * D_ + c] = (__bf16)y;
  }
}

// ---------------------------------------------------------------------------
extern "C" void kernel_launch(void* const* d_in, const int* in_sizes, int n_in,
                              void* d_out, int out_size, void* d_ws, size_t ws_size,
                              hipStream_t stream) {
  (void)in_sizes; (void)n_in; (void)out_size; (void)ws_size;

  const float* x     = (const float*)d_in[0];
  const float* alibi = (const float*)d_in[1];
  const float* Wq = (const float*)d_in[2];  const float* bq = (const float*)d_in[3];
  const float* Wk = (const float*)d_in[4];  const float* bk = (const float*)d_in[5];
  const float* Wv = (const float*)d_in[6];  const float* bv = (const float*)d_in[7];
  const float* Wo = (const float*)d_in[8];  const float* bo = (const float*)d_in[9];
  const float* W1 = (const float*)d_in[10]; const float* b1 = (const float*)d_in[11];
  const float* W2 = (const float*)d_in[12]; const float* b2 = (const float*)d_in[13];
  const float* g1 = (const float*)d_in[14]; const float* be1 = (const float*)d_in[15];
  const float* g2 = (const float*)d_in[16]; const float* be2 = (const float*)d_in[17];

  char* w = (char*)d_ws;
  auto alloc = [&](size_t bytes) -> char* {
    char* p = w;
    w += (bytes + 255) & ~(size_t)255;
    return p;
  };

  __bf16* xb    = (__bf16*)alloc((size_t)M_ * D_ * 2);
  __bf16* WqT   = (__bf16*)alloc((size_t)D_ * D_ * 2);
  __bf16* WkT   = (__bf16*)alloc((size_t)D_ * D_ * 2);
  __bf16* WvT   = (__bf16*)alloc((size_t)D_ * D_ * 2);
  __bf16* WoT   = (__bf16*)alloc((size_t)D_ * D_ * 2);
  __bf16* W1T   = (__bf16*)alloc((size_t)D_ * DFF_ * 2);
  __bf16* W2T   = (__bf16*)alloc((size_t)D_ * DFF_ * 2);
  __bf16* Qb    = (__bf16*)alloc((size_t)M_ * D_ * 2);
  __bf16* Kb    = (__bf16*)alloc((size_t)M_ * D_ * 2);
  __bf16* Vt    = (__bf16*)alloc((size_t)M_ * D_ * 2);
  __bf16* attnB = (__bf16*)alloc((size_t)M_ * D_ * 2);
  __bf16* y1b   = (__bf16*)alloc((size_t)M_ * D_ * 2);
  __bf16* h1    = (__bf16*)alloc((size_t)M_ * DFF_ * 2);
  float*  tmp   = (float*)alloc((size_t)M_ * D_ * 4);
  float*  y1f   = (float*)alloc((size_t)M_ * D_ * 4);

  dim3 tb(32, 8);

  // precision prep: x and all weights -> bf16 (weights transposed for WMMA B)
  cvt_bf16_kernel<<<(M_ * D_) / 256, 256, 0, stream>>>(x, xb, M_ * D_);
  transpose_bf16_kernel<<<dim3(D_ / 32, D_ / 32), tb, 0, stream>>>(Wq, WqT, D_, D_);
  transpose_bf16_kernel<<<dim3(D_ / 32, D_ / 32), tb, 0, stream>>>(Wk, WkT, D_, D_);
  transpose_bf16_kernel<<<dim3(D_ / 32, D_ / 32), tb, 0, stream>>>(Wv, WvT, D_, D_);
  transpose_bf16_kernel<<<dim3(D_ / 32, D_ / 32), tb, 0, stream>>>(Wo, WoT, D_, D_);
  transpose_bf16_kernel<<<dim3(DFF_ / 32, D_ / 32), tb, 0, stream>>>(W1, W1T, D_, DFF_);
  transpose_bf16_kernel<<<dim3(D_ / 32, DFF_ / 32), tb, 0, stream>>>(W2, W2T, DFF_, D_);

  // QKV projections (Q pre-scaled by 1/sqrt(HD)=0.125; V stored transposed)
  dim3 gD(D_ / 64, M_ / 128);
  gemm_bf16_kernel<<<gD, 256, 0, stream>>>(xb, WqT, bq, nullptr, Qb, M_, D_, D_, 1, 0.125f);
  gemm_bf16_kernel<<<gD, 256, 0, stream>>>(xb, WkT, bk, nullptr, Kb, M_, D_, D_, 1, 1.0f);
  gemm_bf16_kernel<<<gD, 256, 0, stream>>>(xb, WvT, bv, nullptr, Vt, M_, D_, D_, 2, 1.0f);

  // flash attention
  attn_kernel<<<B_ * H_ * (S_ / 64), 128, 0, stream>>>(Qb, Kb, Vt, alibi, attnB);

  // output projection + residual, LN1
  gemm_bf16_kernel<<<gD, 256, 0, stream>>>(attnB, WoT, bo, x, tmp, M_, D_, D_, 3, 1.0f);
  ln_kernel<<<M_, 256, 0, stream>>>(tmp, g1, be1, y1f, y1b);

  // FFN
  dim3 gF(DFF_ / 64, M_ / 128);
  gemm_bf16_kernel<<<gF, 256, 0, stream>>>(y1b, W1T, b1, nullptr, h1, M_, DFF_, D_, 4, 1.0f);
  gemm_bf16_kernel<<<gD, 256, 0, stream>>>(h1, W2T, b2, y1f, tmp, M_, D_, DFF_, 3, 1.0f);
  ln_kernel<<<M_, 256, 0, stream>>>(tmp, g2, be2, (float*)d_out, nullptr);
}
